// LocalStridedBlockSparseAttnInference_73297911874194
// MI455X (gfx1250) — matrix operand order repaired and checked by reading, not compiled, not run on hardware
//
#include <hip/hip_runtime.h>
#include <hip/hip_bf16.h>
#include <cmath>

typedef __attribute__((ext_vector_type(16))) _Float16 v16h;
typedef __attribute__((ext_vector_type(8)))  _Float16 v8h;
typedef __attribute__((ext_vector_type(4)))  _Float16 v4h;
typedef __attribute__((ext_vector_type(8)))  float    v8f;
typedef __attribute__((ext_vector_type(4)))  unsigned int u32x4;
typedef __attribute__((ext_vector_type(8)))  int          i32x8;
typedef __attribute__((ext_vector_type(4)))  int          i32x4;

#define B_    2
#define SEQ_  2048
#define H_    32
#define HKV_  8
#define D_    128
#define BLK_  64
#define NB_   (SEQ_ / BLK_)
#define GQ_   (H_ / HKV_)

#define QS_STRIDE 136   // 128 + 8 halves = 64 + 4 dwords  (TDM pad: interval 64dw, amount 4dw)
#define VT_STRIDE 72    // 64 + 8 halves  = 32 + 4 dwords  (TDM pad: interval 32dw, amount 4dw)
#define PS_STRIDE 72

// workspace layout (f16 pre-converted copies)
#define KH_BYTES  ((size_t)B_ * SEQ_ * HKV_ * D_ * 2)   //  8 MB
#define VTH_BYTES ((size_t)B_ * SEQ_ * HKV_ * D_ * 2)   //  8 MB
#define QH_BYTES  ((size_t)B_ * SEQ_ * H_   * D_ * 2)   // 32 MB

// ---------------------------------------------------------------------------
// Pre-pass kernels: one-time f32 -> f16 conversion (K/V blocks are otherwise
// re-converted ~50x by their consumer workgroups). V is transposed per
// 64-token block to [B,HKV,NB,D,BLK] so PV B-fragments are contiguous in K.
// ---------------------------------------------------------------------------
__global__ void cvt_f32_to_f16(const float* __restrict__ src, _Float16* __restrict__ dst,
                               float scale) {
    const size_t i = (size_t)blockIdx.x * blockDim.x + threadIdx.x;   // one float4
    const float4 f = ((const float4*)src)[i];
    v4h h;
    h[0] = (_Float16)(f.x * scale); h[1] = (_Float16)(f.y * scale);
    h[2] = (_Float16)(f.z * scale); h[3] = (_Float16)(f.w * scale);
    ((v4h*)dst)[i] = h;
}

__global__ __launch_bounds__(128)
void cvt_v_transpose(const float* __restrict__ v, _Float16* __restrict__ dst) {
    __shared__ __align__(16) _Float16 T[D_ * VT_STRIDE];
    const int tid = threadIdx.x;
    const int nb  = blockIdx.x % NB_;
    const int hk  = (blockIdx.x / NB_) % HKV_;
    const int b   = blockIdx.x / (NB_ * HKV_);
    const float* vp = v + (((size_t)b * SEQ_ + (size_t)nb * BLK_) * HKV_ + hk) * D_;
    #pragma unroll
    for (int it = 0; it < 16; ++it) {
        int idx = it * 128 + tid;
        int r = idx >> 5, c4 = idx & 31;
        const float4 f = *(const float4*)(vp + (size_t)r * (HKV_ * D_) + c4 * 4);
        T[(c4 * 4 + 0) * VT_STRIDE + r] = (_Float16)f.x;
        T[(c4 * 4 + 1) * VT_STRIDE + r] = (_Float16)f.y;
        T[(c4 * 4 + 2) * VT_STRIDE + r] = (_Float16)f.z;
        T[(c4 * 4 + 3) * VT_STRIDE + r] = (_Float16)f.w;
    }
    __syncthreads();
    _Float16* op = dst + ((((size_t)b * HKV_ + hk) * NB_ + nb) * D_) * BLK_;
    #pragma unroll
    for (int it = 0; it < 16; ++it) {            // 128 rows x 16 v4h each
        int idx = it * 128 + tid;
        int d = idx >> 4, c4 = idx & 15;
        v4h h;
        h[0] = T[d * VT_STRIDE + c4 * 4 + 0];
        h[1] = T[d * VT_STRIDE + c4 * 4 + 1];
        h[2] = T[d * VT_STRIDE + c4 * 4 + 2];
        h[3] = T[d * VT_STRIDE + c4 * 4 + 3];
        *(v4h*)(op + d * BLK_ + c4 * 4) = h;
    }
}

// ---------------------------------------------------------------------------
// TDM: 2-D tensor_load_to_lds (f16, tile == tensor, LDS row padding so WMMA
// fragment loads are bank-conflict free). 6-arg builtin on this toolchain.
// ---------------------------------------------------------------------------
__device__ __forceinline__ void tdm_load_2d(const _Float16* lds_dst, const _Float16* gsrc,
                                            int tile_cols, int tile_rows,
                                            int row_stride_elems, int pad_interval_code) {
    const unsigned long long ga = (unsigned long long)(uintptr_t)gsrc;
    u32x4 g0;
    g0[0] = 1u;                                               // count=1, user descriptor
    g0[1] = (unsigned)(uintptr_t)lds_dst;                     // LDS byte offset
    g0[2] = (unsigned)ga;                                     // global_addr[31:0]
    g0[3] = (unsigned)((ga >> 32) & 0x01FFFFFFu) | (2u << 30);// global_addr[56:32] | type=2
    i32x8 g1;
    g1[0] = (1 << 16)                                         // data_size = 2 bytes
          | (1 << 20)                                         // pad_enable
          | (pad_interval_code << 22)                         // pad every 2^(c+1) dwords
          | (3 << 25);                                        // pad_amount = 4 dwords
    g1[1] = (tile_cols & 0xFFFF) << 16;                       // tensor_dim0[15:0]
    g1[2] = (tile_rows & 0xFFFF) << 16;                       // dim0[31:16]=0 | tensor_dim1[15:0]
    g1[3] = (tile_cols & 0xFFFF) << 16;                       // dim1[31:16]=0 | tile_dim0
    g1[4] = (tile_rows & 0xFFFF);                             // tile_dim1 | tile_dim2=0
    g1[5] = row_stride_elems;                                 // tensor_dim0_stride[31:0]
    g1[6] = 0;
    g1[7] = 0;
    i32x4 z4; z4[0] = z4[1] = z4[2] = z4[3] = 0;              // 2-D: groups 2/3 unused
    i32x8 z8;
    #pragma unroll
    for (int i = 0; i < 8; ++i) z8[i] = 0;
    __builtin_amdgcn_tensor_load_to_lds(g0, g1, z4, z4, z8, 0);
}

__device__ __forceinline__ v16h ld_frag16(const _Float16* p, int gap) {
    v8h lo = *(const v8h*)p;
    v8h hi = *(const v8h*)(p + gap);
    return __builtin_shufflevector(lo, hi, 0,1,2,3,4,5,6,7,8,9,10,11,12,13,14,15);
}

// ---------------------------------------------------------------------------
// Main kernel: one workgroup (4 waves) per (batch, q-head, q-block).
// TDM double-buffers K / V-transposed tiles; WMMA f16 16x16x32 does both GEMMs.
// Softmax runs in base-2 (log2e folded into Q scale); row-sums are computed by
// an extra ones-column WMMA so the l-recurrence rides the O-accumulator path.
// ---------------------------------------------------------------------------
__global__ __launch_bounds__(128)
void bs_flash_attn_f16wmma(const _Float16* __restrict__ qh,
                           const _Float16* __restrict__ kh,
                           const _Float16* __restrict__ vth,
                           const int*      __restrict__ cols,
                           int nnz,
                           float* __restrict__ out)
{
    __shared__ __align__(16) _Float16 Qs[BLK_ * QS_STRIDE];        // Q tile (token, d), pre-scaled
    __shared__ __align__(16) _Float16 Ks[2][BLK_ * QS_STRIDE];     // K tile (token, d), double-buffered
    __shared__ __align__(16) _Float16 Vt[2][D_ * VT_STRIDE];       // V tile transposed (d, token)
    __shared__ __align__(16) _Float16 Ps[4 * 16 * PS_STRIDE];      // per-wave P strips (row, token)

    const int tid  = threadIdx.x;
    const int lane = tid & 31;                  // wave32
    const int wave = tid >> 5;                  // 4 waves
    const int qb   = blockIdx.x % NB_;
    const int h    = (blockIdx.x / NB_) % H_;
    const int b    = blockIdx.x / (NB_ * H_);
    const int hk   = h / GQ_;

    const int c16    = lane & 15;               // N column (B/C/D layouts)
    const int rowoff = (lane >> 4) << 3;        // C/D layout: lanes 16-31 -> M = vgpr+8
    const int ka0    = (lane < 16) ? 0 : 8;     // f16 A layout K base
    const int kb0    = (lane < 16) ? 0 : 8;     // f16 B layout K base
    const int m0     = wave * 16;

    const _Float16* qtile = qh + (((size_t)b * SEQ_ + (size_t)qb * BLK_) * H_ + h) * D_;
    const int* colrow = cols + qb * nnz;

    int col = colrow[0];                         // first entry always valid
    int buf = 0;

    // prologue: start Q + first K/V tiles via the Tensor Data Mover
    if (wave == 0) {
        tdm_load_2d(Qs, qtile, D_, BLK_, H_ * D_, 5);
        const _Float16* kp = kh  + (((size_t)b * SEQ_ + (size_t)col * BLK_) * HKV_ + hk) * D_;
        const _Float16* vp = vth + ((((size_t)b * HKV_ + hk) * NB_ + col) * (size_t)D_) * BLK_;
        tdm_load_2d(&Ks[0][0], kp, D_,   BLK_, HKV_ * D_, 5);
        tdm_load_2d(&Vt[0][0], vp, BLK_, D_,   BLK_,      4);
    }

    // register-built "ones column" B fragment: column c16==0 is 1.0, rest 0.
    v16h ones_b;
    {
        const _Float16 o = (c16 == 0) ? (_Float16)1.0f : (_Float16)0.0f;
        #pragma unroll
        for (int e = 0; e < 16; ++e) ones_b[e] = o;
    }

    // flash state (slot r: lanes<16 -> row m0+r, lanes>=16 -> row m0+r+8)
    float m_r[8];
    v8f   o_acc[8], lacc;
    #pragma unroll
    for (int r = 0; r < 8; ++r) { m_r[r] = -INFINITY; lacc[r] = 0.f; }
    #pragma unroll
    for (int dt = 0; dt < 8; ++dt)
        #pragma unroll
        for (int r = 0; r < 8; ++r) o_acc[dt][r] = 0.f;

    v16h qfrag[4];                               // loop-invariant Q A-fragments (4 d-slices)
    _Float16* psw = &Ps[wave * 16 * PS_STRIDE];

    for (int t = 0; ; ++t) {
        const int  nextcol = (t + 1 < nnz) ? colrow[t + 1] : -1;
        const bool hasnext = nextcol > col;      // padded entries break monotonicity
        const bool diag    = (col == qb);

        __syncthreads();                         // everyone done reading buf^1
        if (wave == 0) {
            if (hasnext) {                       // prefetch next block into buf^1
                const _Float16* kp = kh  + (((size_t)b * SEQ_ + (size_t)nextcol * BLK_) * HKV_ + hk) * D_;
                const _Float16* vp = vth + ((((size_t)b * HKV_ + hk) * NB_ + nextcol) * (size_t)D_) * BLK_;
                tdm_load_2d(&Ks[buf ^ 1][0], kp, D_,   BLK_, HKV_ * D_, 5);
                tdm_load_2d(&Vt[buf ^ 1][0], vp, BLK_, D_,   BLK_,      4);
                __builtin_amdgcn_s_wait_tensorcnt(2);  // in-order: current tiles complete
            } else {
                __builtin_amdgcn_s_wait_tensorcnt(0);
            }
        }
        __syncthreads();                         // publish TDM-written tiles

        if (t == 0) {                            // Q tile now resident: fragments -> VGPRs once
            #pragma unroll
            for (int ks2 = 0; ks2 < 4; ++ks2)
                qfrag[ks2] = ld_frag16(&Qs[(m0 + c16) * QS_STRIDE + ks2 * 32 + ka0], 16);
        }

        const _Float16* ks = &Ks[buf][0];
        const _Float16* vt = &Vt[buf][0];

        // ---- S = (Q*scale*log2e) @ K^T : per-wave 16x64 strip ----
        v8f sacc[4];
        #pragma unroll
        for (int n = 0; n < 4; ++n) {
            #pragma unroll
            for (int r = 0; r < 8; ++r) sacc[n][r] = 0.f;
            #pragma unroll
            for (int ks2 = 0; ks2 < 4; ++ks2) {
                v16h bf = ld_frag16(&ks[(n * 16 + c16) * QS_STRIDE + ks2 * 32 + kb0], 16);
                sacc[n] = __builtin_amdgcn_wmma_f32_16x16x32_f16(
                    false, qfrag[ks2], false, bf, (short)0, sacc[n], false, false);
            }
        }

        if (diag) {                              // token-level causal mask, diagonal block only
            const int myrow = m0 + rowoff;
            #pragma unroll
            for (int n = 0; n < 4; ++n)
                #pragma unroll
                for (int r = 0; r < 8; ++r)
                    sacc[n][r] = (n * 16 + c16 > myrow + r) ? -INFINITY : sacc[n][r];
        }

        // ---- online softmax (base 2) ----
        float nm[8], alpha[8];
        #pragma unroll
        for (int r = 0; r < 8; ++r) {
            float mx = fmaxf(fmaxf(sacc[0][r], sacc[1][r]), fmaxf(sacc[2][r], sacc[3][r]));
            mx = fmaxf(mx, __shfl_xor(mx, 1));
            mx = fmaxf(mx, __shfl_xor(mx, 2));
            mx = fmaxf(mx, __shfl_xor(mx, 4));
            mx = fmaxf(mx, __shfl_xor(mx, 8));   // stays inside each 16-lane half
            nm[r]    = fmaxf(m_r[r], mx);
            alpha[r] = __builtin_amdgcn_exp2f(m_r[r] - nm[r]);
            m_r[r]   = nm[r];
        }
        #pragma unroll
        for (int n = 0; n < 4; ++n)
            #pragma unroll
            for (int r = 0; r < 8; ++r)
                sacc[n][r] = __builtin_amdgcn_exp2f(sacc[n][r] - nm[r]);
        #pragma unroll
        for (int r = 0; r < 8; ++r) lacc[r] *= alpha[r];   // l rides the same recurrence as O
        #pragma unroll
        for (int dt = 0; dt < 8; ++dt)
            #pragma unroll
            for (int r = 0; r < 8; ++r)
                o_acc[dt][r] *= alpha[r];

        // ---- P: C-layout -> f16 A-layout via per-wave LDS strip ----
        #pragma unroll
        for (int n = 0; n < 4; ++n)
            #pragma unroll
            for (int r = 0; r < 8; ++r)
                psw[(r + rowoff) * PS_STRIDE + n * 16 + c16] = (_Float16)sacc[n][r];

        // ---- O += P @ V ; l += P @ ones ----
        #pragma unroll
        for (int ks2 = 0; ks2 < 2; ++ks2) {
            const int ts = ks2 * 32;
            v16h af = ld_frag16(&psw[c16 * PS_STRIDE + ts + ka0], 16);
            #pragma unroll
            for (int dt = 0; dt < 8; ++dt) {
                v16h bf = ld_frag16(&vt[(dt * 16 + c16) * VT_STRIDE + ts + kb0], 16);
                o_acc[dt] = __builtin_amdgcn_wmma_f32_16x16x32_f16(
                    false, af, false, bf, (short)0, o_acc[dt], false, false);
            }
            lacc = __builtin_amdgcn_wmma_f32_16x16x32_f16(
                false, af, false, ones_b, (short)0, lacc, false, false);
        }

        if (!hasnext) break;
        col = nextcol;
        buf ^= 1;
    }

    // ---- normalize + store (l sits in column 0 of lacc: lanes 0 / 16) ----
    float invl[8];
    #pragma unroll
    for (int r = 0; r < 8; ++r) {
        const float l = __shfl(lacc[r], lane & 16);   // broadcast within 16-lane half
        invl[r] = 1.0f / l;
    }
    float* op = out + (((size_t)b * SEQ_ + (size_t)qb * BLK_) * H_ + h) * D_;
    #pragma unroll
    for (int dt = 0; dt < 8; ++dt)
        #pragma unroll
        for (int r = 0; r < 8; ++r) {
            const int row = m0 + rowoff + r;
            op[(size_t)row * (H_ * D_) + dt * 16 + c16] = o_acc[dt][r] * invl[r];
        }
}

extern "C" void kernel_launch(void* const* d_in, const int* in_sizes, int n_in,
                              void* d_out, int out_size, void* d_ws, size_t ws_size,
                              hipStream_t stream) {
    const float* q    = (const float*)d_in[0];
    const float* k    = (const float*)d_in[1];
    const float* v    = (const float*)d_in[2];
    const int*   cols = (const int*)d_in[3];
    // d_in[4] (layout_mask) unused: padding detected via strictly-increasing cols.
    const int nnz = in_sizes[3] / NB_;

    // workspace: f16 K | f16 V-transposed | f16 Q*scale*log2e  (48 MB total)
    char* ws = (char*)d_ws;
    _Float16* kh  = (_Float16*)(ws);
    _Float16* vth = (_Float16*)(ws + KH_BYTES);
    _Float16* qh  = (_Float16*)(ws + KH_BYTES + VTH_BYTES);

    // 1/sqrt(128) * log2(e): base-2 softmax folded into the Q scale
    const float scale = 0.08838834764831845f * 1.44269504088896340f;

    {   // K: elementwise convert  (1,048,576 float4)
        const size_t nf4 = (size_t)B_ * SEQ_ * HKV_ * D_ / 4;
        cvt_f32_to_f16<<<dim3(nf4 / 256), dim3(256), 0, stream>>>(k, kh, 1.0f);
    }
    {   // V: per-block transpose to [B,HKV,NB,D,BLK]
        cvt_v_transpose<<<dim3(B_ * HKV_ * NB_), dim3(128), 0, stream>>>(v, vth);
    }
    {   // Q: elementwise convert with scale folded in
        const size_t nf4 = (size_t)B_ * SEQ_ * H_ * D_ / 4;
        cvt_f32_to_f16<<<dim3(nf4 / 256), dim3(256), 0, stream>>>(q, qh, scale);
    }

    bs_flash_attn_f16wmma<<<dim3(B_ * H_ * NB_), dim3(128), 0, stream>>>(
        qh, kh, vth, cols, nnz, (float*)d_out);
}
